// MaskedAttention_75746043232739
// MI455X (gfx1250) — matrix operand order repaired and checked by reading, not compiled, not run on hardware
//
#include <hip/hip_runtime.h>

// ---------------------------------------------------------------------------
// Performer/RPE masked attention for MI455X (gfx1250), f16 WMMA + f32 accum.
// B=16, L=576, H=12, D=64, M=8.  Dominant op: per-(b,h) GEMM 576x576 * 576x512.
// A (mask) tile staged in LDS via async global->LDS copies (double buffered);
// B (d1t) fragments loaded direct from global (unique per wave).
// ---------------------------------------------------------------------------

#define B_   16
#define L_   576
#define H_   12
#define D_   64
#define M_   8
#define C_   (M_ * D_)      // 512
#define RATIO 0.35355339059327373f   // 1/sqrt(M)
#define NUM_STAB 1e-3f

#define A_STRIDE 40         // f16 per A-tile LDS row (80 B: 16B-aligned, conflict-free)

typedef __attribute__((ext_vector_type(16))) _Float16 v16h;
typedef __attribute__((ext_vector_type(4)))  _Float16 v4h;
typedef __attribute__((ext_vector_type(8)))  float    v8f;
typedef __attribute__((ext_vector_type(4)))  int      v4i;

typedef __attribute__((address_space(1))) v4i* glb_b128_t;   // global int4*
typedef __attribute__((address_space(3))) v4i* lds_b128_t;   // LDS int4*

union FragH { uint4 q[2]; v16h h; };

#if defined(__has_builtin)
#if __has_builtin(__builtin_amdgcn_global_load_async_to_lds_b128)
#define HAVE_ASYNC_LDS 1
#endif
#endif

// stage one 16B chunk of the 64x32 f16 A tile into LDS (256 threads = 4 KB)
__device__ __forceinline__ void stage_a_tile(const _Float16* __restrict__ Abase,
                                             _Float16* s_buf, int i0, int j0, int t) {
  int row = t >> 2, qj = t & 3;                       // 4 x 16B chunks per row
  const _Float16* g = Abase + (size_t)(i0 + row) * L_ + j0 + qj * 8;
  _Float16* l = s_buf + row * A_STRIDE + qj * 8;
#ifdef HAVE_ASYNC_LDS
  __builtin_amdgcn_global_load_async_to_lds_b128((glb_b128_t)g, (lds_b128_t)l, 0, 0);
#else
  *reinterpret_cast<uint4*>(l) = *reinterpret_cast<const uint4*>(g);
#endif
}

__device__ __forceinline__ void async_wait_barrier() {
#ifdef HAVE_ASYNC_LDS
#if __has_builtin(__builtin_amdgcn_s_wait_asynccnt)
  __builtin_amdgcn_s_wait_asynccnt(0);
#else
  asm volatile("s_wait_asynccnt 0x0" ::: "memory");
#endif
#endif
  __syncthreads();
}

// ---------------- mask f32 -> f16 ------------------------------------------
__global__ __launch_bounds__(256) void conv_mask_f16(const float* __restrict__ mask,
                                                     _Float16* __restrict__ maskh) {
  int tid = blockIdx.x * 256 + threadIdx.x;     // one float4 per thread
  const float4 m4 = reinterpret_cast<const float4*>(mask)[tid];
  v4h o; o.x = (_Float16)m4.x; o.y = (_Float16)m4.y; o.z = (_Float16)m4.z; o.w = (_Float16)m4.w;
  reinterpret_cast<v4h*>(maskh)[tid] = o;
}

// ---------------- msum[h][j] = sum_i mask[h][i][j]; zero ks_sum ------------
__global__ __launch_bounds__(256) void prep_msum(const float* __restrict__ mask,
                                                 float* __restrict__ msum,
                                                 float* __restrict__ kss) {
  int tid = blockIdx.x * 256 + threadIdx.x;
  if (tid < B_ * H_ * M_) kss[tid] = 0.f;
  if (tid >= H_ * L_) return;
  int h = tid / L_, j = tid % L_;
  const float* mp = mask + (size_t)h * L_ * L_ + j;
  float s = 0.f;
  for (int i = 0; i < L_; ++i) s += mp[(size_t)i * L_];
  msum[tid] = s;
}

// ---------------- q' = relu(ratio * q . proj) + eps ------------------------
__global__ __launch_bounds__(256) void prep_qprime(const float* __restrict__ qin,
                                                   const float* __restrict__ proj,
                                                   float* __restrict__ qp) {
  __shared__ float s_proj[M_ * D_];
  int t = threadIdx.x;
  s_proj[t] = proj[t];
  s_proj[t + 256] = proj[t + 256];
  __syncthreads();
  int tid = blockIdx.x * 256 + t;               // (b*L+l)*H + h
  const float* qrow = qin + (size_t)tid * D_;
  float acc[M_] = {};
#pragma unroll 4
  for (int d = 0; d < D_; d += 4) {
    float4 qq = *reinterpret_cast<const float4*>(qrow + d);
#pragma unroll
    for (int m = 0; m < M_; ++m) {
      const float* pm = &s_proj[m * D_ + d];
      acc[m] += qq.x * pm[0] + qq.y * pm[1] + qq.z * pm[2] + qq.w * pm[3];
    }
  }
  float* o = qp + (size_t)tid * M_;
#pragma unroll
  for (int m = 0; m < M_; ++m) o[m] = fmaxf(acc[m] * RATIO, 0.f) + NUM_STAB;
}

// ---------------- k', d1t[b][h][c][j] (f16, transposed), ks_sum ------------
__global__ __launch_bounds__(256) void prep_d1t(const float* __restrict__ kin,
                                                const float* __restrict__ vin,
                                                const float* __restrict__ proj,
                                                const float* __restrict__ msum,
                                                float* __restrict__ kss,
                                                _Float16* __restrict__ d1t) {
  __shared__ float s_proj[M_ * D_];
  __shared__ float s_v[64 * 65];      // padded: stride 65 -> conflict free
  __shared__ float s_kp[64 * 9];      // padded: stride 9
  __shared__ float s_ksp[M_];

  int bid = blockIdx.x;
  int it  = bid % 9;
  int h   = (bid / 9) % H_;
  int b   = bid / (9 * H_);
  int l0  = it * 64;
  int t   = threadIdx.x;

  s_proj[t] = proj[t];
  s_proj[t + 256] = proj[t + 256];
  if (t < M_) s_ksp[t] = 0.f;
  // load v tile [64 l][64 d]
#pragma unroll
  for (int r = 0; r < 4; ++r) {
    int idx = t + 256 * r;            // float4 units, 1024 total
    int l = idx >> 4, d4 = idx & 15;
    float4 vv = *reinterpret_cast<const float4*>(
        vin + ((((size_t)b * L_ + l0 + l) * H_ + h) << 6) + d4 * 4);
    float* dst = &s_v[l * 65 + d4 * 4];
    dst[0] = vv.x; dst[1] = vv.y; dst[2] = vv.z; dst[3] = vv.w;
  }
  __syncthreads();

  if (t < 64) {
    int l = l0 + t;
    const float* krow = kin + (((size_t)b * L_ + l) * H_ + h) * D_;
    float acc[M_] = {};
#pragma unroll 4
    for (int d = 0; d < D_; d += 4) {
      float4 kk = *reinterpret_cast<const float4*>(krow + d);
#pragma unroll
      for (int m = 0; m < M_; ++m) {
        const float* pm = &s_proj[m * D_ + d];
        acc[m] += kk.x * pm[0] + kk.y * pm[1] + kk.z * pm[2] + kk.w * pm[3];
      }
    }
    float mj = msum[h * L_ + l];
#pragma unroll
    for (int m = 0; m < M_; ++m) {
      float kp = fmaxf(acc[m] * RATIO, 0.f) + NUM_STAB;
      s_kp[t * 9 + m] = kp;
      atomicAdd(&s_ksp[m], mj * kp);
    }
  }
  __syncthreads();
  if (t < M_) atomicAdd(&kss[((size_t)b * H_ + h) * M_ + t], s_ksp[t]);

  // write d1t[c][l] = k'[l][c/64] * v[l][c%64], c-major so GEMM B-frags are
  // contiguous along j.  Stores coalesced along l (128B rows of f16).
  _Float16* dbase = d1t + (((size_t)b * H_ + h) * C_) * L_ + l0;
#pragma unroll 8
  for (int cit = 0; cit < 128; ++cit) {
    int idx = t + 256 * cit;          // 0..32767
    int c = idx >> 6, l = idx & 63;
    float val = s_kp[l * 9 + (c >> 6)] * s_v[l * 65 + (c & 63)];
    dbase[(size_t)c * L_ + l] = (_Float16)val;
  }
}

// ---------------- fused GEMM (mask x d1) + q'-contraction + normalize ------
// Grid: (b, h, 64-row tile) = 16*12*9 blocks, 256 threads = 8 waves.
// Wave w owns columns c = w*64 .. w*64+63  ==>  feature m = w exactly.
__global__ __launch_bounds__(256, 1) void gemm_attn(const _Float16* __restrict__ maskh,
                                                    const _Float16* __restrict__ d1t,
                                                    const float* __restrict__ qp,
                                                    const float* __restrict__ kss,
                                                    float* __restrict__ out) {
  __shared__ float s_av[64 * 64];
  __shared__ float s_qp[64 * M_];
  __shared__ float s_nrm[64];
  __shared__ float s_kss[M_];
  __shared__ _Float16 s_a[2][64 * A_STRIDE] __attribute__((aligned(16)));

  int bid = blockIdx.x;
  int itile = bid % 9;
  int h = (bid / 9) % H_;
  int b = bid / (9 * H_);
  int i0 = itile * 64;

  int t    = threadIdx.x;
  int w    = t >> 5;        // wave id == feature m
  int lane = t & 31;
  int kh   = lane >> 4;     // half-wave selector for K striping
  int ln   = lane & 15;

  const _Float16* Abase = maskh + (size_t)h * L_ * L_;
  const _Float16* Bbase = d1t + (((size_t)b * H_ + h) * C_ + (size_t)w * 64) * L_;

  // prologue: zero av, stage q'/ks_sum, kick off async copy of first A tile
  stage_a_tile(Abase, s_a[0], i0, 0, t);
  for (int z = t; z < 64 * 64; z += 256) s_av[z] = 0.f;
#pragma unroll
  for (int r = 0; r < 2; ++r) {
    int idx = t + 256 * r;            // 512 q' values for this tile
    int i = idx >> 3, m = idx & 7;
    s_qp[idx] = qp[(((size_t)b * L_ + i0 + i) * H_ + h) * M_ + m];
  }
  if (t < M_) s_kss[t] = kss[((size_t)b * H_ + h) * M_ + t];
  async_wait_barrier();

  v8f acc[4][4] = {};

  for (int step = 0; step < L_ / 32; ++step) {
    int j0 = step * 32;
    // overlap: fetch next A tile into the other buffer while computing
    if (step + 1 < L_ / 32)
      stage_a_tile(Abase, s_a[(step + 1) & 1], i0, j0 + 32, t);

    FragH a[4], bf[4];
    // A (16x32 f16) from LDS: lane ln = row, e0..7 -> K=kh*8+e, e8..15 -> K=16+kh*8+e
    const _Float16* sa = s_a[step & 1];
#pragma unroll
    for (int r = 0; r < 4; ++r) {
      const _Float16* p = sa + (16 * r + ln) * A_STRIDE + kh * 8;
      a[r].q[0] = *reinterpret_cast<const uint4*>(p);
      a[r].q[1] = *reinterpret_cast<const uint4*>(p + 16);
    }
    // B (32x16 f16) direct from global: lane ln = col n, e0..7 -> K=kh*8+e, e8..15 -> K=16+kh*8+e
#pragma unroll
    for (int qi = 0; qi < 4; ++qi) {
      const _Float16* p = Bbase + (size_t)(16 * qi + ln) * L_ + j0 + kh * 8;
      bf[qi].q[0] = *reinterpret_cast<const uint4*>(p);
      bf[qi].q[1] = *reinterpret_cast<const uint4*>(p + 16);
    }
#pragma unroll
    for (int r = 0; r < 4; ++r)
#pragma unroll
      for (int qi = 0; qi < 4; ++qi)
        acc[r][qi] = __builtin_amdgcn_wmma_f32_16x16x32_f16(
            false, a[r].h, false, bf[qi].h, (short)0, acc[r][qi], false, false);

    async_wait_barrier();             // next A buffer ready; all waves past reads
  }

  // epilogue: av[i][d] += q'[i][m=w] * Dhat[i][w*64+d], reduced across waves
#pragma unroll
  for (int r = 0; r < 4; ++r) {
#pragma unroll
    for (int qi = 0; qi < 4; ++qi) {
      int d = 16 * qi + ln;
#pragma unroll
      for (int v = 0; v < 8; ++v) {
        int il = 16 * r + v + 8 * kh;      // C-layout: lanes16-31 -> M=v+8
        atomicAdd(&s_av[il * 64 + d], s_qp[il * M_ + w] * acc[r][qi][v]);
      }
    }
  }
  __syncthreads();

  if (t < 64) {
    float n = 0.f;
#pragma unroll
    for (int m = 0; m < M_; ++m) n += s_qp[t * M_ + m] * s_kss[m];
    s_nrm[t] = n;
  }
  __syncthreads();

  float* obase = out + (((size_t)b * L_ + i0) * H_ + h) * D_;
#pragma unroll
  for (int oit = 0; oit < 16; ++oit) {
    int idx = t + 256 * oit;
    int i = idx >> 6, d = idx & 63;
    obase[(size_t)i * H_ * D_ + d] = s_av[idx] / s_nrm[i];
  }
}

// ---------------------------------------------------------------------------
extern "C" void kernel_launch(void* const* d_in, const int* in_sizes, int n_in,
                              void* d_out, int out_size, void* d_ws, size_t ws_size,
                              hipStream_t stream) {
  (void)in_sizes; (void)n_in; (void)out_size; (void)ws_size;
  const float* q    = (const float*)d_in[0];
  const float* k    = (const float*)d_in[1];
  const float* v    = (const float*)d_in[2];
  const float* proj = (const float*)d_in[3];
  const float* mask = (const float*)d_in[4];
  float* out = (float*)d_out;

  char* ws = (char*)d_ws;
  size_t off = 0;
  auto carve = [&](size_t bytes) -> void* {
    off = (off + 255) & ~(size_t)255;
    void* p = ws + off;
    off += bytes;
    return p;
  };
  float*    qp    = (float*)   carve((size_t)B_ * L_ * H_ * M_ * 4);   // 3.5 MB
  float*    kss   = (float*)   carve((size_t)B_ * H_ * M_ * 4);
  float*    msum  = (float*)   carve((size_t)H_ * L_ * 4);
  _Float16* maskh = (_Float16*)carve((size_t)H_ * L_ * L_ * 2);        // 8 MB
  _Float16* d1t   = (_Float16*)carve((size_t)B_ * H_ * C_ * L_ * 2);   // 113 MB

  conv_mask_f16<<<(H_ * L_ * L_) / 1024, 256, 0, stream>>>(mask, maskh);
  prep_msum   <<<(H_ * L_ + 255) / 256, 256, 0, stream>>>(mask, msum, kss);
  prep_qprime <<<(B_ * L_ * H_) / 256, 256, 0, stream>>>(q, proj, qp);
  prep_d1t    <<<B_ * H_ * 9, 256, 0, stream>>>(k, v, proj, msum, kss, d1t);
  gemm_attn   <<<B_ * H_ * 9, 256, 0, stream>>>(maskh, d1t, qp, kss, out);
}
